// Attention_19198503813415
// MI455X (gfx1250) — compile-verified, matrix operand order checked by reading, not run on hardware
//
#include <hip/hip_runtime.h>

// Problem constants (match reference)
#define BB   2
#define NNq  2048
#define DIMM 1024
#define HH   16
#define HDv  64
#define D3   (3*DIMM)        // 3072
#define MTOT (BB*NNq)        // 4096

// LDS row pitches (elements): data + 16B TDM pad -> conflict-free banks
#define PITCH64 72           // 64-elem rows + 8 pad
#define PITCH32 40           // 32-elem rows + 8 pad

typedef __attribute__((ext_vector_type(16))) __bf16 v16bf;
typedef __attribute__((ext_vector_type(8)))  float  v8f;
typedef __attribute__((ext_vector_type(4)))  unsigned int tdm_u32x4;
typedef __attribute__((ext_vector_type(8)))  int          tdm_i32x8;
typedef __attribute__((ext_vector_type(4)))  int          tdm_i32x4;

union FragAB { uint4 q[2]; v16bf v; };

// round-to-nearest-even fp32 -> bf16 bits
__device__ __forceinline__ unsigned short f2bf(float f) {
  unsigned u = __float_as_uint(f);
  return (unsigned short)((u + 0x7FFFu + ((u >> 16) & 1u)) >> 16);
}

// 16x32 bf16 A/B fragment: per lane-half g, words 0-3 hold k=8g..8g+7 and
// words 4-7 hold k=16+8g..16+8g+7 -> two aligned b128 loads.
__device__ __forceinline__ v16bf load_frag(const unsigned short* base, int off0, int off1) {
  FragAB f;
  f.q[0] = *(const uint4*)(base + off0);
  f.q[1] = *(const uint4*)(base + off1);
  return f.v;
}

// D# group1[0] pad bits: pad_enable | pad_interval code | pad_amount code
__device__ __forceinline__ unsigned pad_bits(unsigned intervalCode, unsigned amountCode) {
  return (1u << 20) | (intervalCode << 22) | (amountCode << 25);
}

// ---------------------------------------------------------------------------
// TDM: D# for a 2D bf16 tile (tile0 elems/row, tile1 rows, row stride in
// elems) -> tensor_load_to_lds. Packing per CDNA5 ISA.
// Tensor dims set huge (2^30) so OOB clipping can never trigger. padbits
// optionally inserts LDS padding every pad_interval DWORDs (load only).
// ---------------------------------------------------------------------------
__device__ __forceinline__ void tdm_load_2d(unsigned lds_addr, const void* gaddr,
                                            unsigned tile0, unsigned tile1,
                                            unsigned long long stride0,
                                            unsigned padbits) {
  const unsigned tdim0 = 1u << 30, tdim1 = 1u << 30;
  unsigned long long ga = (unsigned long long)(uintptr_t)gaddr;
  tdm_u32x4 g0 = {};
  g0[0] = 1u;                                            // count=1 (valid, user)
  g0[1] = lds_addr;                                      // LDS byte address
  g0[2] = (unsigned)(ga & 0xFFFFFFFFu);                  // global_addr[31:0]
  g0[3] = (unsigned)((ga >> 32) & 0x01FFFFFFu) | (2u << 30); // addr[56:32] | type=2
  tdm_i32x8 g1 = {};
  g1[0] = (int)(0x00010000u | padbits);                  // data_size=1 (2B) | pad
  g1[1] = (int)((tdim0 & 0xFFFFu) << 16);                // tensor_dim0[15:0]
  g1[2] = (int)((tdim0 >> 16) | ((tdim1 & 0xFFFFu) << 16));
  g1[3] = (int)((tdim1 >> 16) | (tile0 << 16));          // tile_dim0
  g1[4] = (int)(tile1 & 0xFFFFu);                        // tile_dim1 (tile_dim2=0)
  g1[5] = (int)(stride0 & 0xFFFFFFFFu);                  // tensor_dim0_stride
  g1[6] = (int)((stride0 >> 32) & 0xFFFFu);
  g1[7] = 0;
  tdm_i32x4 gz = {0, 0, 0, 0};                           // 2D: groups 2/3 unused
#if __has_include(<hip/amd_detail/amd_gfx1250_TDM.h>)
  tdm_i32x8 gz8 = {0, 0, 0, 0, 0, 0, 0, 0};
  __builtin_amdgcn_tensor_load_to_lds(g0, g1, gz, gz, gz8, 0);
#else
  __builtin_amdgcn_tensor_load_to_lds(g0, g1, gz, gz, 0);
#endif
}

// ---------------------------------------------------------------------------
// Kernel 1: fp32 -> bf16; transpose weights so B-operand rows are contiguous.
// ---------------------------------------------------------------------------
__global__ void convert_pack(const float* __restrict__ x,
                             const float* __restrict__ wqkv,
                             const float* __restrict__ wproj,
                             unsigned short* __restrict__ xb,
                             unsigned short* __restrict__ wqkvT,
                             unsigned short* __restrict__ wprojT) {
  int idx = blockIdx.x * blockDim.x + threadIdx.x;
  const int NX  = BB*NNq*DIMM;   // 4M
  const int NW1 = DIMM*D3;       // 3M
  if (idx < NX) {
    xb[idx] = f2bf(x[idx]);
  } else if (idx < NX + NW1) {
    int i = idx - NX;
    int d = i / DIMM, c = i % DIMM;
    wqkvT[i] = f2bf(wqkv[c * D3 + d]);
  } else {
    int i = idx - NX - NW1;
    int d = i / DIMM, c = i % DIMM;
    wprojT[i] = f2bf(wproj[c * DIMM + d]);
  }
}

// ---------------------------------------------------------------------------
// Shared GEMM body: one block = 8 waves x 32 rows = 256 rows, 64 columns.
// BOTH operand tiles TDM-staged into LDS (A: 256x64, B: 64x64 per 64-k
// phase), double-buffered on TENSORcnt; 16 WMMAs per phase, all fragments
// transient LDS reads (no register pipelines, no WAR-hazard movs).
// ---------------------------------------------------------------------------
#define GEMM_PROLOGUE(ABASE, WTBASE)                                          \
  __shared__ __align__(16) unsigned short shA[2][256*PITCH64];  /* 2x36KB */  \
  __shared__ __align__(16) unsigned short shB[2][64*PITCH64];   /* 2x9KB  */  \
  int w    = threadIdx.x >> 5;                                                \
  int lane = threadIdx.x & 31;                                                \
  int lh = lane & 15, g = lane >> 4;                                          \
  int mbase = (blockIdx.x & 15) * 256;                                        \
  int m0 = mbase + w * 32;                                                    \
  int c0 = (blockIdx.x >> 4) * 64;                                            \
  const unsigned short* atile = (ABASE) + (size_t)mbase * DIMM;               \
  const unsigned short* wtile = (WTBASE) + (size_t)c0 * DIMM;                 \
  v8f acc0[4] = {{},{},{},{}};                                                \
  v8f acc1[4] = {{},{},{},{}};                                                \
  if (threadIdx.x < 32) {                                                     \
    tdm_load_2d((unsigned)(uintptr_t)&shA[0][0], atile, 64u, 256u,            \
                (unsigned long long)DIMM, pad_bits(4u, 3u));                  \
    tdm_load_2d((unsigned)(uintptr_t)&shB[0][0], wtile, 64u, 64u,             \
                (unsigned long long)DIMM, pad_bits(4u, 3u));                  \
  }                                                                           \
  for (int i = 0, kk = 0; kk < DIMM; kk += 64, ++i) {                         \
    int bsel = i & 1;                                                         \
    __syncthreads();                      /* prior reads of other buf done */ \
    if (threadIdx.x < 32) {                                                   \
      if (kk + 64 < DIMM) {                                                   \
        tdm_load_2d((unsigned)(uintptr_t)&shA[1 - bsel][0], atile + kk + 64,  \
                    64u, 256u, (unsigned long long)DIMM, pad_bits(4u, 3u));   \
        tdm_load_2d((unsigned)(uintptr_t)&shB[1 - bsel][0], wtile + kk + 64,  \
                    64u, 64u, (unsigned long long)DIMM, pad_bits(4u, 3u));    \
        __builtin_amdgcn_s_wait_tensorcnt(2);  /* in-order: cur bufs landed */\
      } else {                                                                \
        __builtin_amdgcn_s_wait_tensorcnt(0);                                 \
      }                                                                       \
    }                                                                         \
    __syncthreads();                      /* release: tiles visible */        \
    _Pragma("unroll")                                                         \
    for (int kc = 0; kc < 64; kc += 32) {                                     \
      v16bf a0 = load_frag(&shA[bsel][(w*32 + lh) * PITCH64],                 \
                           kc + 8*g, kc + 16 + 8*g);                          \
      v16bf a1 = load_frag(&shA[bsel][(w*32 + 16 + lh) * PITCH64],            \
                           kc + 8*g, kc + 16 + 8*g);                          \
      _Pragma("unroll")                                                       \
      for (int t = 0; t < 4; ++t) {                                           \
        v16bf b = load_frag(&shB[bsel][(16*t + lh) * PITCH64],                \
                            kc + 8*g, kc + 16 + 8*g);                         \
        acc0[t] = __builtin_amdgcn_wmma_f32_16x16x32_bf16(false, a0, false, b,\
                                              (short)0, acc0[t], false, false); \
        acc1[t] = __builtin_amdgcn_wmma_f32_16x16x32_bf16(false, a1, false, b,\
                                              (short)0, acc1[t], false, false); \
      }                                                                       \
    }                                                                         \
  }

// ---------------------------------------------------------------------------
// Kernel 2: QKV GEMM [4096,1024]x[1024,3072]; epilogue scatters Q (scaled,
// [B,H,N,HD]), K ([B,H,N,HD]), V^T ([B,H,HD,N]) in bf16.
// ---------------------------------------------------------------------------
__device__ __forceinline__ void qkv_store(float val, int m, int d,
                                          unsigned short* qb, unsigned short* kb,
                                          unsigned short* vTb) {
  int p = d >> 10, rem = d & 1023;
  int h = rem >> 6, hd = rem & 63;
  int b_ = m >> 11, n_ = m & (NNq - 1);
  int bh = b_ * HH + h;
  if (p == 0)      qb [(bh*NNq + n_)*HDv + hd] = f2bf(val * 0.125f);  // HD^-0.5
  else if (p == 1) kb [(bh*NNq + n_)*HDv + hd] = f2bf(val);
  else             vTb[(bh*HDv + hd)*NNq + n_] = f2bf(val);
}

__global__ __launch_bounds__(256, 1)
void qkv_gemm(const unsigned short* __restrict__ xb,
              const unsigned short* __restrict__ wT,
              unsigned short* __restrict__ qb,
              unsigned short* __restrict__ kb,
              unsigned short* __restrict__ vTb) {
  GEMM_PROLOGUE(xb, wT)
#pragma unroll
  for (int t = 0; t < 4; ++t) {
    int d = c0 + 16*t + lh;
#pragma unroll
    for (int r = 0; r < 8; ++r) {
      qkv_store(acc0[t][r], m0 + r + 8*g,      d, qb, kb, vTb);
      qkv_store(acc1[t][r], m0 + 16 + r + 8*g, d, qb, kb, vTb);
    }
  }
}

// ---------------------------------------------------------------------------
// Kernel 4: proj GEMM [4096,1024]x[1024,1024] + bias, fp32 out.
// ---------------------------------------------------------------------------
__global__ __launch_bounds__(256, 1)
void proj_gemm(const unsigned short* __restrict__ ab,
               const unsigned short* __restrict__ wTp,
               const float* __restrict__ bias,
               float* __restrict__ out) {
  GEMM_PROLOGUE(ab, wTp)
#pragma unroll
  for (int t = 0; t < 4; ++t) {
    int d = c0 + 16*t + lh;
    float bv = bias[d];
#pragma unroll
    for (int r = 0; r < 8; ++r) {
      out[(m0 + r + 8*g) * DIMM + d]      = acc0[t][r] + bv;
      out[(m0 + 16 + r + 8*g) * DIMM + d] = acc1[t][r] + bv;
    }
  }
}

// ---------------------------------------------------------------------------
// Kernel 3: flash attention. 8 waves/block share one (b,h); K/V tiles TDM-
// staged into LDS with DMA-inserted row padding (bank-conflict-free),
// double-buffered on TENSORcnt. Per 32-key step: 4 WMMAs for S, online
// softmax (shfl_xor row reductions), P via padded per-wave LDS slab,
// 4 WMMAs for O += P V.
// ---------------------------------------------------------------------------
__global__ void flash_attn(const unsigned short* __restrict__ qb,
                           const unsigned short* __restrict__ kb,
                           const unsigned short* __restrict__ vTb,
                           unsigned short* __restrict__ aout) {
  __shared__ __align__(16) unsigned short shK[2][32*PITCH64];   // 2 x 4.5KB
  __shared__ __align__(16) unsigned short shV[2][HDv*PITCH32];  // 2 x 5KB
  __shared__ __align__(16) unsigned short plds[8][16*PITCH32];  // 10KB

  int w    = threadIdx.x >> 5;
  int lane = threadIdx.x & 31;
  int lh = lane & 15, g = lane >> 4;
  int bh = blockIdx.x >> 4;                 // 0..31 (b*H + h)
  int mt = (blockIdx.x & 15) * 8 + w;       // Q row tile 0..127

  const unsigned short* kbase = kb  + (size_t)bh * NNq * HDv;
  const unsigned short* vbase = vTb + (size_t)bh * HDv * NNq;

  const unsigned short* qrow = qb + ((size_t)bh * NNq + mt*16 + lh) * HDv;
  v16bf aq0 = load_frag(qrow,      8*g, 16 + 8*g);
  v16bf aq1 = load_frag(qrow, 32 + 8*g, 48 + 8*g);

  v8f o[4] = {{},{},{},{}};
  float rowm[8], rowl[8];
#pragma unroll
  for (int r = 0; r < 8; ++r) { rowm[r] = -1e30f; rowl[r] = 0.f; }

  bool issuer = (threadIdx.x < 32);         // wave 0 drives the TDM
  if (issuer) {
    // K tile: 32 rows x 64 elems, contiguous rows; pad 16B per 128B row.
    tdm_load_2d((unsigned)(uintptr_t)&shK[0][0], kbase, 64u, 32u,
                (unsigned long long)HDv, pad_bits(4u, 3u));
    // V^T tile: 64 rows x 32 elems, row stride N; pad 16B per 64B row.
    tdm_load_2d((unsigned)(uintptr_t)&shV[0][0], vbase, 32u, (unsigned)HDv,
                (unsigned long long)NNq, pad_bits(3u, 3u));
  }

  for (int i = 0, nn = 0; nn < NNq; nn += 32, ++i) {
    int bsel = i & 1;
    __syncthreads();                        // prior reads of buf(1-bsel) done
    if (issuer) {
      if (nn + 32 < NNq) {
        tdm_load_2d((unsigned)(uintptr_t)&shK[1 - bsel][0],
                    kbase + (size_t)(nn + 32)*HDv, 64u, 32u,
                    (unsigned long long)HDv, pad_bits(4u, 3u));
        tdm_load_2d((unsigned)(uintptr_t)&shV[1 - bsel][0], vbase + (nn + 32),
                    32u, (unsigned)HDv, (unsigned long long)NNq, pad_bits(3u, 3u));
        __builtin_amdgcn_s_wait_tensorcnt(2);  // in-order: current buffer landed
      } else {
        __builtin_amdgcn_s_wait_tensorcnt(0);
      }
    }
    __syncthreads();                        // release: LDS tiles visible

    // ---- S = Q K^T for 32 keys (two 16-wide tiles), K frags from LDS ----
    v8f s[2];
#pragma unroll
    for (int u = 0; u < 2; ++u) {
      const unsigned short* krow = &shK[bsel][(16*u + lh) * PITCH64];
      v16bf bk0 = load_frag(krow,      8*g, 16 + 8*g);
      v16bf bk1 = load_frag(krow, 32 + 8*g, 48 + 8*g);
      v8f t = {};
      t = __builtin_amdgcn_wmma_f32_16x16x32_bf16(false, aq0, false, bk0, (short)0, t, false, false);
      t = __builtin_amdgcn_wmma_f32_16x16x32_bf16(false, aq1, false, bk1, (short)0, t, false, false);
      s[u] = t;
    }

    // ---- online softmax: rows in regs (r+8g), cols striped over 16 lanes ----
#pragma unroll
    for (int r = 0; r < 8; ++r) {
      float mx = fmaxf(s[0][r], s[1][r]);
      mx = fmaxf(mx, __shfl_xor(mx, 1, 32));
      mx = fmaxf(mx, __shfl_xor(mx, 2, 32));
      mx = fmaxf(mx, __shfl_xor(mx, 4, 32));
      mx = fmaxf(mx, __shfl_xor(mx, 8, 32));
      float newm = fmaxf(rowm[r], mx);
      float corr = __expf(rowm[r] - newm);
      float p0 = __expf(s[0][r] - newm);
      float p1 = __expf(s[1][r] - newm);
      float ps = p0 + p1;
      ps += __shfl_xor(ps, 1, 32);
      ps += __shfl_xor(ps, 2, 32);
      ps += __shfl_xor(ps, 4, 32);
      ps += __shfl_xor(ps, 8, 32);
      rowl[r] = rowl[r] * corr + ps;
      rowm[r] = newm;
#pragma unroll
      for (int t2 = 0; t2 < 4; ++t2) o[t2][r] = o[t2][r] * corr;
      int row = r + 8*g;
      plds[w][row*PITCH32 +      lh] = f2bf(p0);
      plds[w][row*PITCH32 + 16 + lh] = f2bf(p1);
    }

    // ---- O += P V : P from per-wave LDS slab, V frags from staged LDS ----
    v16bf ap = load_frag(&plds[w][0], lh*PITCH32 + 8*g, lh*PITCH32 + 16 + 8*g);
#pragma unroll
    for (int t2 = 0; t2 < 4; ++t2) {
      const unsigned short* vrow = &shV[bsel][(16*t2 + lh) * PITCH32];
      v16bf bv = load_frag(vrow, 8*g, 16 + 8*g);
      o[t2] = __builtin_amdgcn_wmma_f32_16x16x32_bf16(false, ap, false, bv,
                                                      (short)0, o[t2], false, false);
    }
  }

  // ---- epilogue: normalize, write bf16 [B,N,C] for proj GEMM ----
  int b_ = bh >> 4, h = bh & 15;
#pragma unroll
  for (int t2 = 0; t2 < 4; ++t2) {
#pragma unroll
    for (int r = 0; r < 8; ++r) {
      int row = mt*16 + r + 8*g;
      float val = o[t2][r] / rowl[r];
      aout[((size_t)b_ * NNq + row) * DIMM + h * HDv + 16*t2 + lh] = f2bf(val);
    }
  }
}

// ---------------------------------------------------------------------------
extern "C" void kernel_launch(void* const* d_in, const int* in_sizes, int n_in,
                              void* d_out, int out_size, void* d_ws, size_t ws_size,
                              hipStream_t stream) {
  (void)in_sizes; (void)n_in; (void)out_size; (void)ws_size;
  const float* x      = (const float*)d_in[0];
  const float* w_qkv  = (const float*)d_in[1];
  const float* w_proj = (const float*)d_in[2];
  const float* b_proj = (const float*)d_in[3];
  float* out = (float*)d_out;

  char* ws = (char*)d_ws;
  size_t off = 0;
  unsigned short* xb     = (unsigned short*)(ws + off); off += (size_t)BB*NNq*DIMM*2;   // 8 MB
  unsigned short* wqkvT  = (unsigned short*)(ws + off); off += (size_t)D3*DIMM*2;       // 6 MB
  unsigned short* wprojT = (unsigned short*)(ws + off); off += (size_t)DIMM*DIMM*2;     // 2 MB
  unsigned short* qb     = (unsigned short*)(ws + off); off += (size_t)BB*HH*NNq*HDv*2; // 8 MB
  unsigned short* kb     = (unsigned short*)(ws + off); off += (size_t)BB*HH*NNq*HDv*2; // 8 MB
  unsigned short* vTb    = (unsigned short*)(ws + off); off += (size_t)BB*HH*HDv*NNq*2; // 8 MB
  unsigned short* aout   = (unsigned short*)(ws + off); off += (size_t)BB*NNq*DIMM*2;   // 8 MB

  convert_pack<<<(BB*NNq*DIMM + DIMM*D3 + DIMM*DIMM) / 256, 256, 0, stream>>>(
      x, w_qkv, w_proj, xb, wqkvT, wprojT);

  // 16 row-groups (256 rows each) x 48 col-groups (64 cols each)
  qkv_gemm<<<16*48, 256, 0, stream>>>(xb, wqkvT, qb, kb, vTb);

  // 32 (b,h) x 16 blocks x 8 waves (one 16-row Q tile each)
  flash_attn<<<32*16, 256, 0, stream>>>(qb, kb, vTb, aout);

  // 16 row-groups x 16 col-groups
  proj_gemm<<<16*16, 256, 0, stream>>>(aout, wprojT, b_proj, out);
}
